// ODELSTM_88252987998722
// MI455X (gfx1250) — matrix operand order repaired and checked by reading, not compiled, not run on hardware
//
#include <hip/hip_runtime.h>
#include <hip/hip_bf16.h>

#define D_    128
#define H_    2048
#define B_    256
#define T_    8
#define OBS_  128
#define G4H_  8192
#define TWO_LOGC 1.8378770664093453f

#define LDS_PAD_ROW 72          // 64 data bf16 + 8 pad  (144 B, 16B-aligned)
#define AS_ELEMS (64 * LDS_PAD_ROW)
#define WS_ELEMS (128 * LDS_PAD_ROW)

typedef __attribute__((ext_vector_type(16))) __bf16 v16bf;
typedef __attribute__((ext_vector_type(8)))  __bf16 v8bf;
typedef __attribute__((ext_vector_type(8)))  float  v8f;

static __device__ __forceinline__ __bf16 f2bf(float x) {
  union { float f; unsigned u; } v; v.f = x;
  unsigned ru = v.u + 0x7FFFu + ((v.u >> 16) & 1u);   // round-to-nearest-even
  unsigned short h = (unsigned short)(ru >> 16);
  __bf16 o; __builtin_memcpy(&o, &h, 2);
  return o;
}

// ---------------------------------------------------------------- converts
__global__ void cvt_bf16_kernel(const float* __restrict__ src,
                                __bf16* __restrict__ dst, int n) {
  int stride = (int)(gridDim.x * blockDim.x);
  for (int i = blockIdx.x * blockDim.x + threadIdx.x; i < n; i += stride)
    dst[i] = f2bf(src[i]);
}

static __device__ __forceinline__ v16bf pack16(v8bf lo, v8bf hi) {
  v16bf a;
#pragma unroll
  for (int i = 0; i < 8; ++i) { a[i] = lo[i]; a[i + 8] = hi[i]; }
  return a;
}

// --------------------------------------------------- async global -> LDS
static __device__ __forceinline__ unsigned lds_off_of(const void* p) {
  // generic LDS pointer = {aperture, lds_offset}; low 32 bits = LDS offset
  return (unsigned)(unsigned long long)p;
}

static __device__ __forceinline__ void async_cp16(unsigned ldsOff, const __bf16* g) {
  asm volatile("global_load_async_to_lds_b128 %0, %1, off"
               :: "v"(ldsOff), "v"(g) : "memory");
}

static __device__ __forceinline__ void wait_async0() {
  asm volatile("s_wait_asynccnt 0x0" ::: "memory");
}

// Copy one 64-k chunk of the block tile into LDS:
//   A: 64 rows x 64 k (8 KB), W: 128 rows x 64 k (16 KB)  => 1536 x 16B pieces,
//   6 async instructions per thread; each wave's 32 lanes cover 32 contiguous
//   16B pieces (block boundaries are 32-aligned => no divergence).
static __device__ __forceinline__ void prefetch_tile(
    __bf16* As, __bf16* Ws,
    const __bf16* Ag, const __bf16* Wg, int K, int tid) {
#pragma unroll
  for (int q = 0; q < 6; ++q) {
    int c = q * 256 + tid;
    if (c < 512) {                       // A region: 64 rows * 8 chunks
      int row = c >> 3, ch = c & 7;
      async_cp16(lds_off_of(As + row * LDS_PAD_ROW + ch * 8),
                 Ag + (size_t)row * K + ch * 8);
    } else {                             // W region: 128 rows * 8 chunks
      int c2 = c - 512;
      int row = c2 >> 3, ch = c2 & 7;
      async_cp16(lds_off_of(Ws + row * LDS_PAD_ROW + ch * 8),
                 Wg + (size_t)row * K + ch * 8);
    }
  }
}

// ------------------------------------------------- pipelined block GEMM
// Accumulates C(64x128 block tile) += A(64,K) @ W(128,K)^T with double-
// buffered async LDS staging. Wave (wm,wn) owns the 16x64 subtile.
static __device__ __forceinline__ void gemm_block_acc(
    const __bf16* __restrict__ Ag,   // A + bm*64*K  (global, row stride K)
    const __bf16* __restrict__ Wg,   // W + bn*128*K (global, row stride K)
    int K,
    __bf16 (*As)[AS_ELEMS], __bf16 (*Ws)[WS_ELEMS],
    int tid, int wm, int wn, int half, int r,
    v8f& a0, v8f& a1, v8f& a2, v8f& a3) {
  const int KC = K >> 6;
  prefetch_tile(As[0], Ws[0], Ag, Wg, K, tid);
  wait_async0();
  __syncthreads();
  for (int kc = 0; kc < KC; ++kc) {
    const int cur = kc & 1;
    if (kc + 1 < KC)
      prefetch_tile(As[cur ^ 1], Ws[cur ^ 1],
                    Ag + (kc + 1) * 64, Wg + (kc + 1) * 64, K, tid);
    const __bf16* Abuf = As[cur];
    const __bf16* Wbuf = Ws[cur];
#pragma unroll
    for (int s = 0; s < 2; ++s) {       // two 32-k sub-chunks
      const __bf16* arow = Abuf + (wm * 16 + r) * LDS_PAD_ROW + s * 32 + half * 16;
      v8bf alo = *(const v8bf*)(arow);
      v8bf ahi = *(const v8bf*)(arow + 8);
      v16bf a = pack16(alo, ahi);
      const __bf16* wbase = Wbuf + (wn * 64 + r) * LDS_PAD_ROW + s * 32 + half * 16;
      v16bf b0 = *(const v16bf*)(wbase + 0 * 16 * LDS_PAD_ROW);
      v16bf b1 = *(const v16bf*)(wbase + 1 * 16 * LDS_PAD_ROW);
      v16bf b2 = *(const v16bf*)(wbase + 2 * 16 * LDS_PAD_ROW);
      v16bf b3 = *(const v16bf*)(wbase + 3 * 16 * LDS_PAD_ROW);
      a0 = __builtin_amdgcn_wmma_f32_16x16x32_bf16(false, a, false, b0, (short)0, a0, false, false);
      a1 = __builtin_amdgcn_wmma_f32_16x16x32_bf16(false, a, false, b1, (short)0, a1, false, false);
      a2 = __builtin_amdgcn_wmma_f32_16x16x32_bf16(false, a, false, b2, (short)0, a2, false, false);
      a3 = __builtin_amdgcn_wmma_f32_16x16x32_bf16(false, a, false, b3, (short)0, a3, false, false);
    }
    if (kc + 1 < KC) {   // block-uniform condition
      wait_async0();
      __syncthreads();   // next buffer ready; all waves done reading cur
    }
  }
}

// ------------------------------------------------------------ GEMM kernel
// C(M,N) = A(M,K) @ W(N,K)^T + bias; block tile 64x128 (waves 4x2 of 16x64).
//  EPI 0: outB = bf16(tanh(v))   EPI 1: hF += DT*v; outB = bf16(hF)
//  EPI 2: outB = bf16(relu(v))   EPI 3: outF = v
template <int EPI>
__global__ __launch_bounds__(256) void gemm_bf16_kernel(
    const __bf16* __restrict__ A, const __bf16* __restrict__ W,
    const float* __restrict__ bias,
    float* __restrict__ outF, __bf16* __restrict__ outB,
    float* __restrict__ hF, int M, int N, int K) {
  __shared__ __bf16 As[2][AS_ELEMS];
  __shared__ __bf16 Ws[2][WS_ELEMS];
  const int tid  = threadIdx.x;
  const int wave = tid >> 5;
  const int lane = tid & 31;
  const int half = lane >> 4;
  const int r    = lane & 15;
  const int wm = wave >> 1;            // 0..3
  const int wn = wave & 1;             // 0..1
  const int nbn = N >> 7;              // blocks along N
  const int bm = blockIdx.x / nbn;
  const int bn = blockIdx.x % nbn;

  v8f acc0 = {}, acc1 = {}, acc2 = {}, acc3 = {};
  gemm_block_acc(A + (size_t)bm * 64 * K, W + (size_t)bn * 128 * K, K,
                 As, Ws, tid, wm, wn, half, r, acc0, acc1, acc2, acc3);

  const int mBase = bm * 64 + wm * 16 + half * 8;
  v8f accs[4] = {acc0, acc1, acc2, acc3};
#pragma unroll
  for (int nn = 0; nn < 4; ++nn) {
    const int n = bn * 128 + wn * 64 + nn * 16 + r;
    const float bv = bias[n];
#pragma unroll
    for (int i = 0; i < 8; ++i) {
      const size_t o = (size_t)(mBase + i) * N + n;
      float v = accs[nn][i] + bv;
      if (EPI == 0) {
        outB[o] = f2bf(tanhf(v));
      } else if (EPI == 1) {
        float nh = hF[o] + 0.05f * v;
        hF[o] = nh;
        outB[o] = f2bf(nh);
      } else if (EPI == 2) {
        outB[o] = f2bf(fmaxf(v, 0.0f));
      } else {
        outF[o] = v;
      }
    }
  }
}

// ------------------------------------------------------------ gates GEMM
// gates(128,8192) = Xo(128,128)@Wih^T + Hsel(128,2048)@Whh^T + bih + bhh
__global__ __launch_bounds__(256) void gates_kernel(
    const __bf16* __restrict__ Xo, const __bf16* __restrict__ Hsel,
    const __bf16* __restrict__ Wih, const __bf16* __restrict__ Whh,
    const float* __restrict__ bih, const float* __restrict__ bhh,
    float* __restrict__ gates) {
  __shared__ __bf16 As[2][AS_ELEMS];
  __shared__ __bf16 Ws[2][WS_ELEMS];
  const int tid  = threadIdx.x;
  const int wave = tid >> 5;
  const int lane = tid & 31;
  const int half = lane >> 4;
  const int r    = lane & 15;
  const int wm = wave >> 1;
  const int wn = wave & 1;
  const int bm = blockIdx.x >> 6;      // nbn = 8192/128 = 64
  const int bn = blockIdx.x & 63;

  v8f acc0 = {}, acc1 = {}, acc2 = {}, acc3 = {};
  gemm_block_acc(Xo + (size_t)bm * 64 * D_, Wih + (size_t)bn * 128 * D_, D_,
                 As, Ws, tid, wm, wn, half, r, acc0, acc1, acc2, acc3);
  __syncthreads();   // all waves finished segment 1 buffers before re-staging
  gemm_block_acc(Hsel + (size_t)bm * 64 * H_, Whh + (size_t)bn * 128 * H_, H_,
                 As, Ws, tid, wm, wn, half, r, acc0, acc1, acc2, acc3);

  const int mBase = bm * 64 + wm * 16 + half * 8;
  v8f accs[4] = {acc0, acc1, acc2, acc3};
#pragma unroll
  for (int nn = 0; nn < 4; ++nn) {
    const int n = bn * 128 + wn * 64 + nn * 16 + r;
    const float bv = bih[n] + bhh[n];
#pragma unroll
    for (int i = 0; i < 8; ++i)
      gates[(size_t)(mBase + i) * G4H_ + n] = accs[nn][i] + bv;
  }
}

// ------------------------------------------------------------- elementwise
__global__ void gather_kernel(const __bf16* __restrict__ hB,
                              const int* __restrict__ idx,
                              __bf16* __restrict__ hsel) {
  int g = blockIdx.x * blockDim.x + threadIdx.x;
  int j = g >> 11, d = g & 2047;
  hsel[(size_t)j * H_ + d] = hB[(size_t)idx[j] * H_ + d];
}

__global__ __launch_bounds__(256) void loss_kernel(
    const float* __restrict__ p, const float* __restrict__ Xo,
    const float* __restrict__ Mo, float* __restrict__ acc) {
  int g = blockIdx.x * blockDim.x + threadIdx.x;
  int j = g >> 7, d = g & 127;
  float mean = p[(size_t)j * 256 + d];
  float lv   = p[(size_t)j * 256 + 128 + d];
  float m    = Mo[(size_t)j * 128 + d];
  float err  = (Xo[(size_t)j * 128 + d] - mean) * expf(-0.5f * lv);
  float l    = 0.5f * (err * err + lv + TWO_LOGC) * m;
  __shared__ float sl[256];
  __shared__ float sm[256];
  int t = threadIdx.x;
  sl[t] = l; sm[t] = m;
  __syncthreads();
  for (int o2 = 128; o2 > 0; o2 >>= 1) {
    if (t < o2) { sl[t] += sl[t + o2]; sm[t] += sm[t + o2]; }
    __syncthreads();
  }
  if (t == 0) { atomicAdd(&acc[0], sl[0]); atomicAdd(&acc[1], sm[0]); }
}

__global__ void lstm_kernel(const float* __restrict__ gates,
                            const int* __restrict__ idx,
                            float* __restrict__ cF, float* __restrict__ hF,
                            __bf16* __restrict__ hB) {
  int g = blockIdx.x * blockDim.x + threadIdx.x;
  int j = g >> 11, d = g & 2047;
  int row = idx[j];
  const float* gr = gates + (size_t)j * G4H_;
  float ig = gr[d], fg = gr[H_ + d], gg = gr[2 * H_ + d], og = gr[3 * H_ + d];
  float co = cF[(size_t)row * H_ + d];
  float si = 1.0f / (1.0f + expf(-ig));
  float sf = 1.0f / (1.0f + expf(-fg));
  float so = 1.0f / (1.0f + expf(-og));
  float cn = sf * co + si * tanhf(gg);
  float hn = so * tanhf(cn);
  cF[(size_t)row * H_ + d] = cn;
  hF[(size_t)row * H_ + d] = hn;
  hB[(size_t)row * H_ + d] = f2bf(hn);
}

__global__ void finalize_kernel(const float* __restrict__ acc, float* __restrict__ out) {
  out[0] = acc[0] / acc[1];
}

// ------------------------------------------------------------------- host
extern "C" void kernel_launch(void* const* d_in, const int* in_sizes, int n_in,
                              void* d_out, int out_size, void* d_ws, size_t ws_size,
                              hipStream_t stream) {
  (void)in_sizes; (void)n_in; (void)out_size; (void)ws_size;
  const float* X    = (const float*)d_in[0];
  const float* Mm   = (const float*)d_in[1];
  const float* W1   = (const float*)d_in[2];
  const float* b1   = (const float*)d_in[3];
  const float* W2   = (const float*)d_in[4];
  const float* b2   = (const float*)d_in[5];
  const float* Wih  = (const float*)d_in[6];
  const float* Whh  = (const float*)d_in[7];
  const float* bih  = (const float*)d_in[8];
  const float* bhh  = (const float*)d_in[9];
  const float* Wp1  = (const float*)d_in[10];
  const float* bp1  = (const float*)d_in[11];
  const float* Wp2  = (const float*)d_in[12];
  const float* bp2  = (const float*)d_in[13];
  const int*   bidx = (const int*)d_in[16];

  char* ws = (char*)d_ws;
  size_t off = 0;
  auto alloc = [&](size_t bytes) -> char* {
    char* p = ws + off;
    off += (bytes + 255) & ~(size_t)255;
    return p;
  };
  __bf16* W1b   = (__bf16*)alloc((size_t)H_ * H_ * 2);
  __bf16* W2b   = (__bf16*)alloc((size_t)H_ * H_ * 2);
  __bf16* Wp1b  = (__bf16*)alloc((size_t)H_ * H_ * 2);
  __bf16* Wp2b  = (__bf16*)alloc((size_t)2 * D_ * H_ * 2);
  __bf16* Wihb  = (__bf16*)alloc((size_t)G4H_ * D_ * 2);
  __bf16* Whhb  = (__bf16*)alloc((size_t)G4H_ * H_ * 2);
  __bf16* Xb    = (__bf16*)alloc((size_t)T_ * OBS_ * D_ * 2);
  float*  hF    = (float*) alloc((size_t)B_ * H_ * 4);
  __bf16* hB    = (__bf16*)alloc((size_t)B_ * H_ * 2);
  float*  cF    = (float*) alloc((size_t)B_ * H_ * 4);
  __bf16* tmpB  = (__bf16*)alloc((size_t)B_ * H_ * 2);
  __bf16* hselB = (__bf16*)alloc((size_t)OBS_ * H_ * 2);
  __bf16* P1B   = (__bf16*)alloc((size_t)OBS_ * H_ * 2);
  float*  pF    = (float*) alloc((size_t)OBS_ * 2 * D_ * 4);
  float*  gatesF= (float*) alloc((size_t)OBS_ * G4H_ * 4);
  float*  lossA = (float*) alloc(2 * sizeof(float));

  cvt_bf16_kernel<<<4096, 256, 0, stream>>>(W1,  W1b,  H_ * H_);
  cvt_bf16_kernel<<<4096, 256, 0, stream>>>(W2,  W2b,  H_ * H_);
  cvt_bf16_kernel<<<4096, 256, 0, stream>>>(Wp1, Wp1b, H_ * H_);
  cvt_bf16_kernel<<<1024, 256, 0, stream>>>(Wp2, Wp2b, 2 * D_ * H_);
  cvt_bf16_kernel<<<2048, 256, 0, stream>>>(Wih, Wihb, G4H_ * D_);
  cvt_bf16_kernel<<<4096, 256, 0, stream>>>(Whh, Whhb, G4H_ * H_);
  cvt_bf16_kernel<<<512,  256, 0, stream>>>(X,   Xb,   T_ * OBS_ * D_);

  hipMemsetAsync(hF, 0, (size_t)B_ * H_ * 4, stream);
  hipMemsetAsync(hB, 0, (size_t)B_ * H_ * 2, stream);
  hipMemsetAsync(cF, 0, (size_t)B_ * H_ * 4, stream);
  hipMemsetAsync(lossA, 0, 2 * sizeof(float), stream);

  double t = 0.0;
  for (int i = 0; i < T_; ++i) {
    float ot = (float)((double)(i + 1) * 0.1);   // matches np.arange*0.1 astype f32
    while (t < (double)ot - 1e-9) {              // replicate reference step schedule
      gemm_bf16_kernel<0><<<64, 256, 0, stream>>>(hB, W1b, b1, nullptr, tmpB, nullptr,
                                                  B_, H_, H_);
      gemm_bf16_kernel<1><<<64, 256, 0, stream>>>(tmpB, W2b, b2, nullptr, hB, hF,
                                                  B_, H_, H_);
      t += 0.05;
    }
    int s = i * OBS_;
    gather_kernel<<<1024, 256, 0, stream>>>(hB, bidx + s, hselB);
    gemm_bf16_kernel<2><<<32, 256, 0, stream>>>(hselB, Wp1b, bp1, nullptr, P1B, nullptr,
                                                OBS_, H_, H_);
    gemm_bf16_kernel<3><<<4, 256, 0, stream>>>(P1B, Wp2b, bp2, pF, nullptr, nullptr,
                                               OBS_, 2 * D_, H_);
    loss_kernel<<<64, 256, 0, stream>>>(pF, X + (size_t)s * D_, Mm + (size_t)s * D_, lossA);
    gates_kernel<<<128, 256, 0, stream>>>(Xb + (size_t)s * D_, hselB, Wihb, Whhb,
                                          bih, bhh, gatesF);
    lstm_kernel<<<1024, 256, 0, stream>>>(gatesF, bidx + s, cF, hF, hB);
  }
  finalize_kernel<<<1, 1, 0, stream>>>(lossA, (float*)d_out);
}
